// MultiHeadAttention_bias_9732395892965
// MI455X (gfx1250) — matrix operand order repaired and checked by reading, not compiled, not run on hardware
//
#include <hip/hip_runtime.h>
#include <hip/hip_bf16.h>

// ---------------------------------------------------------------------------
// MI455X / gfx1250 fused MHA-with-bias + LayerNorm.
// All GEMMs use v_wmma_f32_16x16x32_f16 (wave32, 16x16 C tiles, f32 accum).
// Every WMMA operand is loaded K-contiguous per lane -> b128 vector loads.
// ---------------------------------------------------------------------------

#define BB   4
#define SS   1024
#define DD   512
#define HH   8
#define DKV  64
#define NROW (BB * SS)          // 4096
#define NEGC (-1000000000.0f)

typedef __attribute__((ext_vector_type(16))) _Float16 v16h;
typedef __attribute__((ext_vector_type(8)))  float    v8f;

// K-index of half-element e (0..15) for lane-half hi, A-matrix 16x32 layout
__device__ __forceinline__ int kmapA(int e, int hi) {
    return ((e < 8) ? 0 : 16) + hi * 8 + (e & 7);
}
// K-index for B-matrix 32x16 layout (same order as A: lanes 0-15: K=0..7,16..23; lanes 16-31: K=8..15,24..31)
__device__ __forceinline__ int kmapB(int e, int hi) {
    return ((e < 8) ? 0 : 16) + hi * 8 + (e & 7);
}

// A fragment (16x32) from row-major [m][k] block, base at (m0,k0), ld = row stride.
// Per lane: two contiguous 16-byte runs -> global_load_b128 x2.
__device__ __forceinline__ v16h frag_a_rowmajor(const _Float16* __restrict__ base, int ld) {
    int lane = threadIdx.x & 31;
    int m = lane & 15, hi = lane >> 4;
    v16h f;
#pragma unroll
    for (int e = 0; e < 16; ++e) f[e] = base[m * ld + kmapA(e, hi)];
    return f;
}

// B fragment (32x16) from K-contiguous storage [n][k] (i.e. the transpose of the
// logical KxN operand), base at (n0,k0), ld = row stride of the [n][k] storage.
// Per lane: two contiguous 16-byte runs -> global_load_b128 x2.
__device__ __forceinline__ v16h frag_b_kcontig(const _Float16* __restrict__ base, int ld) {
    int lane = threadIdx.x & 31;
    int n = lane & 15, hi = lane >> 4;
    v16h f;
#pragma unroll
    for (int e = 0; e < 16; ++e) f[e] = base[n * ld + kmapB(e, hi)];
    return f;
}

__device__ __forceinline__ v8f wmma_f16(v16h a, v16h b, v8f c) {
    return __builtin_amdgcn_wmma_f32_16x16x32_f16(false, a, false, b, (short)0, c,
                                                  false, false);
}

// ---------------------------------------------------------------------------
// fp32 -> f16 conversion (activations)
// ---------------------------------------------------------------------------
__global__ __launch_bounds__(256) void cvt_f32_to_f16(const float* __restrict__ in,
                                                      _Float16* __restrict__ out, int n) {
    int i = blockIdx.x * 256 + threadIdx.x;
    if (i < n) out[i] = (_Float16)in[i];
}

// ---------------------------------------------------------------------------
// fp32 (512x512, [k][n]) -> f16 transposed ([n][k]) via 32x32 LDS tile.
// Coalesced global reads and writes; makes every weight B-fragment K-contiguous.
// ---------------------------------------------------------------------------
__global__ __launch_bounds__(256) void cvt_w_transpose(const float* __restrict__ in,
                                                       _Float16* __restrict__ out) {
    __shared__ _Float16 tile[32][33];
    int k0 = blockIdx.x * 32;            // source row block  (k)
    int n0 = blockIdx.y * 32;            // source col block  (n)
    int tx = threadIdx.x & 31, ty = threadIdx.x >> 5;   // 8 rows/pass, 4 passes
#pragma unroll
    for (int p = 0; p < 4; ++p) {
        int r = ty + p * 8;
        tile[r][tx] = (_Float16)in[(size_t)(k0 + r) * DD + n0 + tx];
    }
    __syncthreads();
#pragma unroll
    for (int p = 0; p < 4; ++p) {
        int r = ty + p * 8;
        out[(size_t)(n0 + r) * DD + k0 + tx] = tile[tx][r];
    }
}

// ---------------------------------------------------------------------------
// Projection GEMM: Y(head-major f16) = X(4096x512 f16) @ W (via Wt[n][k] f16)
// One 16x16 C tile per wave; 4 waves per block.
// ---------------------------------------------------------------------------
__global__ __launch_bounds__(128) void proj_gemm(const _Float16* __restrict__ X,
                                                 const _Float16* __restrict__ Wt,
                                                 _Float16* __restrict__ Yh) {
    int wave = threadIdx.x >> 5;
    int tile = blockIdx.x * 4 + wave;              // 8192 tiles = 256 x 32
    int m0 = (tile >> 5) * 16;
    int n0 = (tile & 31) * 16;
    v8f acc = {};
#pragma unroll 4
    for (int k0 = 0; k0 < DD; k0 += 32) {
        __builtin_prefetch(X + m0 * DD + k0 + 64, 0, 0);   // global_prefetch_b8
        v16h a = frag_a_rowmajor(X + m0 * DD + k0, DD);
        v16h b = frag_b_kcontig(Wt + n0 * DD + k0, DD);
        acc = wmma_f16(a, b, acc);
    }
    int lane = threadIdx.x & 31;
    int n = lane & 15, hi = lane >> 4;
#pragma unroll
    for (int r = 0; r < 8; ++r) {
        int row = m0 + r + 8 * hi;                 // b*S + s
        int col = n0 + n;                          // h*64 + d
        int b = row >> 10, s = row & (SS - 1);
        int h = col >> 6,  d = col & (DKV - 1);
        Yh[(((size_t)(b * HH + h)) * SS + s) * DKV + d] = (_Float16)acc[r];
    }
}

// Same GEMM but writes the result transposed per head: Yt[(b,h,d,s)].
// Used for V so the attention B-fragments (P @ V) are K-contiguous.
__global__ __launch_bounds__(128) void proj_gemm_vt(const _Float16* __restrict__ X,
                                                    const _Float16* __restrict__ Wt,
                                                    _Float16* __restrict__ Yt) {
    int wave = threadIdx.x >> 5;
    int tile = blockIdx.x * 4 + wave;
    int m0 = (tile >> 5) * 16;
    int n0 = (tile & 31) * 16;
    v8f acc = {};
#pragma unroll 4
    for (int k0 = 0; k0 < DD; k0 += 32) {
        __builtin_prefetch(X + m0 * DD + k0 + 64, 0, 0);
        v16h a = frag_a_rowmajor(X + m0 * DD + k0, DD);
        v16h b = frag_b_kcontig(Wt + n0 * DD + k0, DD);
        acc = wmma_f16(a, b, acc);
    }
    int lane = threadIdx.x & 31;
    int n = lane & 15, hi = lane >> 4;
#pragma unroll
    for (int r = 0; r < 8; ++r) {
        int row = m0 + r + 8 * hi;                 // b*S + s
        int col = n0 + n;                          // h*64 + d
        int b = row >> 10, s = row & (SS - 1);
        int h = col >> 6,  d = col & (DKV - 1);
        Yt[(((size_t)(b * HH + h)) * DKV + d) * SS + s] = (_Float16)acc[r];
    }
}

// ---------------------------------------------------------------------------
// Attention: one (b,h, 16-query strip) per block, 4 waves.
//  phase 1: scores(16x1024) = QK^T/8 + masked bias  -> LDS (f16)
//  phase 2: row softmax (fp32 stats), write attn fp32 to d_out, probs f16 to LDS
//  phase 3: context(16x64) = P @ V via WMMA, A from LDS, B from Vt (b128 loads)
// ---------------------------------------------------------------------------
__global__ __launch_bounds__(128) void attn_kernel(
    const _Float16* __restrict__ Qh, const _Float16* __restrict__ Kh,
    const _Float16* __restrict__ Q2h, const _Float16* __restrict__ K2h,
    const _Float16* __restrict__ Vt,
    const unsigned char* __restrict__ am, const unsigned char* __restrict__ bm,
    float* __restrict__ attn_out, _Float16* __restrict__ ctx) {

    __shared__ _Float16 s_p[16][SS];               // 32 KB score/prob strip
    __shared__ float s_max[16], s_inv[16];

    int bh = blockIdx.x;                           // b*H + h
    int b  = bh >> 3, h = bh & 7;
    int q0 = blockIdx.y * 16;
    int wave = threadIdx.x >> 5;
    int lane = threadIdx.x & 31;
    int nl = lane & 15, hi = lane >> 4;

    const _Float16* Qbase  = Qh  + ((size_t)bh * SS + q0) * DKV;
    const _Float16* Q2base = Q2h + ((size_t)bh * SS + q0) * DKV;
    v16h aq0  = frag_a_rowmajor(Qbase, DKV);
    v16h aq1  = frag_a_rowmajor(Qbase + 32, DKV);
    v16h aq20 = frag_a_rowmajor(Q2base, DKV);
    v16h aq21 = frag_a_rowmajor(Q2base + 32, DKV);

    const unsigned char* amB = am + (size_t)b * SS * SS;
    const unsigned char* bmB = bm + (size_t)b * SS * SS;

    // ---- phase 1: scores (K rows are d-contiguous -> b128 B-fragment loads) ----
    for (int kt = wave; kt < SS / 16; kt += 4) {
        int k0 = kt * 16;
        const _Float16* Kbase  = Kh  + ((size_t)bh * SS + k0) * DKV;
        const _Float16* K2base = K2h + ((size_t)bh * SS + k0) * DKV;
        v8f s1 = {};
        s1 = wmma_f16(aq0, frag_b_kcontig(Kbase, DKV), s1);
        s1 = wmma_f16(aq1, frag_b_kcontig(Kbase + 32, DKV), s1);
        v8f sb = {};
        sb = wmma_f16(aq20, frag_b_kcontig(K2base, DKV), sb);
        sb = wmma_f16(aq21, frag_b_kcontig(K2base + 32, DKV), sb);
#pragma unroll
        for (int r = 0; r < 8; ++r) {
            int qr = r + 8 * hi;
            int q  = q0 + qr;
            int k  = k0 + nl;
            float bias = bmB[(size_t)q * SS + k] ? NEGC : sb[r];
            float sc   = amB[(size_t)q * SS + k] ? NEGC : (s1[r] * 0.125f + bias);
            sc = fmaxf(sc, -30000.0f);             // finite in f16; exp() still 0
            s_p[qr][k] = (_Float16)sc;
        }
    }
    __syncthreads();

    // ---- phase 2: softmax ----
    for (int rr = 0; rr < 4; ++rr) {
        int r = wave * 4 + rr;
        float mx = -3.0e38f;
        for (int j = lane; j < SS; j += 32) mx = fmaxf(mx, (float)s_p[r][j]);
#pragma unroll
        for (int off = 16; off; off >>= 1) mx = fmaxf(mx, __shfl_xor(mx, off, 32));
        float sum = 0.0f;
        for (int j = lane; j < SS; j += 32) sum += __expf((float)s_p[r][j] - mx);
#pragma unroll
        for (int off = 16; off; off >>= 1) sum += __shfl_xor(sum, off, 32);
        if (lane == 0) { s_max[r] = mx; s_inv[r] = 1.0f / sum; }
    }
    __syncthreads();

    float* aout = attn_out + ((size_t)bh * SS + q0) * SS;
    for (int idx = threadIdx.x; idx < 16 * SS; idx += 128) {
        int r = idx >> 10, k = idx & (SS - 1);
        float p = __expf((float)s_p[r][k] - s_max[r]) * s_inv[r];
        s_p[r][k] = (_Float16)p;
        aout[(size_t)r * SS + k] = p;
    }
    __syncthreads();

    // ---- phase 3: context = P @ V  (wave -> its 16-wide slice of DV) ----
    int n0 = wave * 16;
    v8f cacc = {};
    for (int k0 = 0; k0 < SS; k0 += 32) {
        v16h a;
#pragma unroll
        for (int e = 0; e < 16; ++e) a[e] = s_p[nl][k0 + kmapA(e, hi)];   // ds_load_b128 x2
        v16h bf = frag_b_kcontig(Vt + ((size_t)bh * DKV + n0) * SS + k0, SS);
        cacc = wmma_f16(a, bf, cacc);
    }
#pragma unroll
    for (int r = 0; r < 8; ++r) {
        int s   = q0 + r + 8 * hi;
        int col = h * DKV + n0 + nl;
        ctx[((size_t)(b * SS + s)) * DD + col] = (_Float16)cacc[r];
    }
}

// ---------------------------------------------------------------------------
// Output GEMM: x = ctx(4096x512 f16) @ Wo (via WoT[n][k]) + residual (fp32)
// ---------------------------------------------------------------------------
__global__ __launch_bounds__(128) void out_gemm(const _Float16* __restrict__ C,
                                                const _Float16* __restrict__ WoT,
                                                const float* __restrict__ residual,
                                                float* __restrict__ xbuf) {
    int wave = threadIdx.x >> 5;
    int tile = blockIdx.x * 4 + wave;
    int m0 = (tile >> 5) * 16;
    int n0 = (tile & 31) * 16;
    v8f acc = {};
#pragma unroll 4
    for (int k0 = 0; k0 < DD; k0 += 32) {
        __builtin_prefetch(C + m0 * DD + k0 + 64, 0, 0);
        v16h a = frag_a_rowmajor(C + m0 * DD + k0, DD);
        v16h b = frag_b_kcontig(WoT + n0 * DD + k0, DD);
        acc = wmma_f16(a, b, acc);
    }
    int lane = threadIdx.x & 31;
    int n = lane & 15, hi = lane >> 4;
#pragma unroll
    for (int r = 0; r < 8; ++r) {
        size_t idx = (size_t)(m0 + r + 8 * hi) * DD + (n0 + n);
        xbuf[idx] = acc[r] + residual[idx];
    }
}

// ---------------------------------------------------------------------------
// LayerNorm over D=512 per row
// ---------------------------------------------------------------------------
__global__ __launch_bounds__(128) void layernorm_kernel(const float* __restrict__ x,
                                                        const float* __restrict__ g,
                                                        const float* __restrict__ bt,
                                                        float* __restrict__ out) {
    __shared__ float wsum[4], wsq[4];
    int row = blockIdx.x;
    const float* xr = x + (size_t)row * DD;
    int wave = threadIdx.x >> 5, lane = threadIdx.x & 31;
    float v[4], lsum = 0.0f, lsq = 0.0f;
#pragma unroll
    for (int i = 0; i < 4; ++i) {
        v[i] = xr[threadIdx.x + i * 128];
        lsum += v[i];
        lsq  += v[i] * v[i];
    }
#pragma unroll
    for (int off = 16; off; off >>= 1) {
        lsum += __shfl_xor(lsum, off, 32);
        lsq  += __shfl_xor(lsq,  off, 32);
    }
    if (lane == 0) { wsum[wave] = lsum; wsq[wave] = lsq; }
    __syncthreads();
    float tsum = wsum[0] + wsum[1] + wsum[2] + wsum[3];
    float tsq  = wsq[0]  + wsq[1]  + wsq[2]  + wsq[3];
    float mu   = tsum * (1.0f / DD);
    float var  = tsq * (1.0f / DD) - mu * mu;
    float rstd = rsqrtf(var + 1e-5f);
#pragma unroll
    for (int i = 0; i < 4; ++i) {
        int c = threadIdx.x + i * 128;
        out[(size_t)row * DD + c] = (v[i] - mu) * rstd * g[c] + bt[c];
    }
}

// ---------------------------------------------------------------------------
// Host-side orchestration
// ---------------------------------------------------------------------------
extern "C" void kernel_launch(void* const* d_in, const int* in_sizes, int n_in,
                              void* d_out, int out_size, void* d_ws, size_t ws_size,
                              hipStream_t stream) {
    const float* input_Q = (const float*)d_in[0];
    const float* input_K = (const float*)d_in[1];
    const float* input_V = (const float*)d_in[2];
    const unsigned char* attn_mask = (const unsigned char*)d_in[3];
    const unsigned char* bias_mask = (const unsigned char*)d_in[4];
    const float* Wq  = (const float*)d_in[5];
    const float* Wk  = (const float*)d_in[6];
    const float* Wq2 = (const float*)d_in[7];
    const float* Wk2 = (const float*)d_in[8];
    const float* Wv  = (const float*)d_in[9];
    const float* Wo  = (const float*)d_in[10];
    const float* ln_g = (const float*)d_in[11];
    const float* ln_b = (const float*)d_in[12];

    const size_t NX = (size_t)NROW * DD;   // 2,097,152 activation elems
    const size_t NW = (size_t)DD * DD;     // 262,144 weight elems

    char* w = (char*)d_ws;
    _Float16* Xq = (_Float16*)w;  w += NX * sizeof(_Float16);
    _Float16* Xk = (_Float16*)w;  w += NX * sizeof(_Float16);
    _Float16* Xv = (_Float16*)w;  w += NX * sizeof(_Float16);
    _Float16* WqT  = (_Float16*)w; w += NW * sizeof(_Float16);
    _Float16* WkT  = (_Float16*)w; w += NW * sizeof(_Float16);
    _Float16* Wq2T = (_Float16*)w; w += NW * sizeof(_Float16);
    _Float16* Wk2T = (_Float16*)w; w += NW * sizeof(_Float16);
    _Float16* WvT  = (_Float16*)w; w += NW * sizeof(_Float16);
    _Float16* WoT  = (_Float16*)w; w += NW * sizeof(_Float16);
    _Float16* Qh  = (_Float16*)w; w += NX * sizeof(_Float16);
    _Float16* Kh  = (_Float16*)w; w += NX * sizeof(_Float16);
    _Float16* Q2h = (_Float16*)w; w += NX * sizeof(_Float16);
    _Float16* K2h = (_Float16*)w; w += NX * sizeof(_Float16);
    _Float16* Vt  = (_Float16*)w; w += NX * sizeof(_Float16);
    _Float16* Ctx = (_Float16*)w; w += NX * sizeof(_Float16);
    float*    Xb  = (float*)w;    w += NX * sizeof(float);

    float* out_final = (float*)d_out;                       // (B,S,D)
    float* attn_out  = (float*)d_out + NX;                  // (B,H,S,S)

    // 1) fp32 -> f16 conversions (weights transposed to K-contiguous layout)
    int gx = (int)((NX + 255) / 256);
    cvt_f32_to_f16<<<gx, 256, 0, stream>>>(input_Q, Xq, (int)NX);
    cvt_f32_to_f16<<<gx, 256, 0, stream>>>(input_K, Xk, (int)NX);
    cvt_f32_to_f16<<<gx, 256, 0, stream>>>(input_V, Xv, (int)NX);
    dim3 tg(DD / 32, DD / 32);
    cvt_w_transpose<<<tg, 256, 0, stream>>>(Wq,  WqT);
    cvt_w_transpose<<<tg, 256, 0, stream>>>(Wk,  WkT);
    cvt_w_transpose<<<tg, 256, 0, stream>>>(Wq2, Wq2T);
    cvt_w_transpose<<<tg, 256, 0, stream>>>(Wk2, Wk2T);
    cvt_w_transpose<<<tg, 256, 0, stream>>>(Wv,  WvT);
    cvt_w_transpose<<<tg, 256, 0, stream>>>(Wo,  WoT);

    // 2) head projections (WMMA GEMMs): 8192 tiles / 4 waves per block
    const int projBlocks = (NROW / 16) * (DD / 16) / 4;     // 2048
    proj_gemm<<<projBlocks, 128, 0, stream>>>(Xq, WqT,  Qh);
    proj_gemm<<<projBlocks, 128, 0, stream>>>(Xk, WkT,  Kh);
    proj_gemm<<<projBlocks, 128, 0, stream>>>(Xq, Wq2T, Q2h);
    proj_gemm<<<projBlocks, 128, 0, stream>>>(Xk, Wk2T, K2h);
    proj_gemm_vt<<<projBlocks, 128, 0, stream>>>(Xv, WvT, Vt);

    // 3) attention core: grid (B*H, S/16)
    attn_kernel<<<dim3(BB * HH, SS / 16), 128, 0, stream>>>(
        Qh, Kh, Q2h, K2h, Vt, attn_mask, bias_mask, attn_out, Ctx);

    // 4) output projection + residual
    out_gemm<<<projBlocks, 128, 0, stream>>>(Ctx, WoT, input_Q, Xb);

    // 5) LayerNorm
    layernorm_kernel<<<NROW, 128, 0, stream>>>(Xb, ln_g, ln_b, out_final);

    (void)in_sizes; (void)n_in; (void)out_size; (void)ws_size;
}
